// MiniGPT4AttentionCam_7919919694021
// MI455X (gfx1250) — compile-verified, matrix operand order checked by reading, not run on hardware
//
#include <hip/hip_runtime.h>
#include <hip/hip_bf16.h>
#include <math.h>

// ---------------------------------------------------------------------------
// MiniGPT4 attention with CAM mask, CDNA5 (gfx1250) WMMA implementation.
// f16 matrix operands, f32 accumulation via v_wmma_f32_16x16x32_f16.
// Async global->LDS staging (ASYNCcnt) for the shared K chunk in attention.
// ---------------------------------------------------------------------------

typedef __attribute__((ext_vector_type(16))) _Float16 v16h;
typedef __attribute__((ext_vector_type(8)))  float    v8f;

#define SEQLEN 4096
#define EMB    2048
#define NHEAD  16
#define HDIM   128
#define KEY_LO 410     // ceil(0.1 * 4096): keys <  KEY_LO are kept
#define KEY_HI 3072    // 4096 - ceil(0.25*4096): keys >= KEY_HI are kept
#define NCHUNK 45      // 13 chunks of 32 cover [0,416), 32 chunks cover [3072,4096)
#define LDSTRIDE 40    // padded LDS row stride in halves (80 B, keeps 16B alignment)
#define KSTRIDE  136   // padded LDS row stride for the K chunk (272 B, 16B-aligned)

union Frag16 { v16h v; uint4 u[2]; _Float16 h[16]; };

__device__ __forceinline__ v8f wmma_f16(const Frag16& a, const Frag16& b, v8f c) {
  // D = A(16x32 f16) * B(32x16 f16) + C(16x16 f32)
  return __builtin_amdgcn_wmma_f32_16x16x32_f16(false, a.v, false, b.v,
                                                (short)0, c, false, false);
}

// Async copy of 64 contiguous bytes global -> LDS (per-lane addresses).
// The instruction offset is added to BOTH the LDS and global addresses
// (CDNA5 ISA), so one base pair covers all four b128s.
__device__ __forceinline__ void async_copy64(unsigned lds_off, const void* gptr) {
  asm volatile(
      "global_load_async_to_lds_b128 %0, %1, off\n\t"
      "global_load_async_to_lds_b128 %0, %1, off offset:16\n\t"
      "global_load_async_to_lds_b128 %0, %1, off offset:32\n\t"
      "global_load_async_to_lds_b128 %0, %1, off offset:48"
      :: "v"(lds_off), "v"((unsigned long long)(uintptr_t)gptr)
      : "memory");
}

__device__ __forceinline__ void wait_async0() {
#if __has_builtin(__builtin_amdgcn_s_wait_asynccnt)
  __builtin_amdgcn_s_wait_asynccnt(0);
#else
  asm volatile("s_wait_asynccnt 0x0" ::: "memory");
#endif
}

// ---------------------------------------------------------------------------
// GEMM: Out[M=4096, N=2048] = A[4096,2048] @ W^T + bias,  W is (N,K) row-major.
// Block: 256 threads = 8 waves; tile 128x128; K-step 32.
// Wave (w&3, w>>2) computes a 32x64 sub-tile = 2x4 WMMA C tiles.
// ---------------------------------------------------------------------------
template<bool A_HALF, bool OUT_F32>
__global__ __launch_bounds__(256) void gemm_wmma(const void* __restrict__ Ap,
                                                 const float* __restrict__ W,
                                                 const float* __restrict__ bias,
                                                 void* __restrict__ Out) {
  __shared__ __align__(16) _Float16 As[128 * LDSTRIDE];
  __shared__ __align__(16) _Float16 Bs[128 * LDSTRIDE];

  const int t    = threadIdx.x;
  const int wave = t >> 5;
  const int lane = t & 31;
  const int l16  = lane & 15;
  const int hi   = lane >> 4;           // half-wave select
  const int mblk = blockIdx.y * 128;
  const int nblk = blockIdx.x * 128;
  const int mw   = (wave & 3) * 32;     // wave M offset inside tile
  const int nw   = (wave >> 2) * 64;    // wave N offset inside tile
  const int srow = t >> 1;              // staging: row 0..127
  const int scol = (t & 1) * 16;        // staging: 16-element half of the K slab
  const int koffs = hi ? 8 : 0;         // A-layout K offset for this half-wave

  v8f acc[8];
#pragma unroll
  for (int i = 0; i < 8; ++i) acc[i] = {};

  for (int k0 = 0; k0 < EMB; k0 += 32) {
    __syncthreads();
    // ---- stage A tile (128 rows x 32 K) into LDS as f16 ----
    if (A_HALF) {
      const _Float16* a = (const _Float16*)Ap + (size_t)(mblk + srow) * EMB + k0 + scol;
      uint4 u0 = ((const uint4*)a)[0];
      uint4 u1 = ((const uint4*)a)[1];
      *(uint4*)(As + srow * LDSTRIDE + scol)     = u0;
      *(uint4*)(As + srow * LDSTRIDE + scol + 8) = u1;
      if (k0 + 32 < EMB) __builtin_prefetch((const void*)(a + 32), 0, 3);
    } else {
      const float* a = (const float*)Ap + (size_t)(mblk + srow) * EMB + k0 + scol;
      union { float4 f4[4]; float s[16]; } fa;
      const float4* a4 = (const float4*)a;
      fa.f4[0] = a4[0]; fa.f4[1] = a4[1]; fa.f4[2] = a4[2]; fa.f4[3] = a4[3];
      Frag16 cv;
#pragma unroll
      for (int i = 0; i < 16; ++i) cv.h[i] = (_Float16)fa.s[i];
      *(uint4*)(As + srow * LDSTRIDE + scol)     = cv.u[0];
      *(uint4*)(As + srow * LDSTRIDE + scol + 8) = cv.u[1];
      if (k0 + 32 < EMB) __builtin_prefetch((const void*)(a + 32), 0, 3);
    }
    // ---- stage B tile: rows are W rows (the N index), 32 K columns ----
    {
      const float* b = W + (size_t)(nblk + srow) * EMB + k0 + scol;
      union { float4 f4[4]; float s[16]; } fb;
      const float4* b4 = (const float4*)b;
      fb.f4[0] = b4[0]; fb.f4[1] = b4[1]; fb.f4[2] = b4[2]; fb.f4[3] = b4[3];
      Frag16 cv;
#pragma unroll
      for (int i = 0; i < 16; ++i) cv.h[i] = (_Float16)fb.s[i];
      *(uint4*)(Bs + srow * LDSTRIDE + scol)     = cv.u[0];
      *(uint4*)(Bs + srow * LDSTRIDE + scol + 8) = cv.u[1];
      if (k0 + 32 < EMB) __builtin_prefetch((const void*)(b + 32), 0, 3);
    }
    __syncthreads();

    // ---- fragments + WMMA ----
    Frag16 afr[2];
#pragma unroll
    for (int i = 0; i < 2; ++i) {
      const _Float16* p = As + (mw + i * 16 + l16) * LDSTRIDE + koffs;
      afr[i].u[0] = *(const uint4*)p;
      afr[i].u[1] = *(const uint4*)(p + 16);
    }
#pragma unroll
    for (int j = 0; j < 4; ++j) {
      Frag16 bfr;
      const _Float16* p = Bs + (nw + j * 16 + l16) * LDSTRIDE + koffs;
      bfr.u[0] = *(const uint4*)p;
      bfr.u[1] = *(const uint4*)(p + 16);
#pragma unroll
      for (int i = 0; i < 2; ++i)
        acc[i * 4 + j] = wmma_f16(afr[i], bfr, acc[i * 4 + j]);
    }
  }

  // ---- epilogue: bias + store (C layout: N = l16, M = r + 8*hi) ----
#pragma unroll
  for (int j = 0; j < 4; ++j) {
    const int gn  = nblk + nw + j * 16 + l16;
    const float bv = bias[gn];
#pragma unroll
    for (int i = 0; i < 2; ++i) {
#pragma unroll
      for (int r = 0; r < 8; ++r) {
        const int gm  = mblk + mw + i * 16 + r + 8 * hi;
        const float v = acc[i * 4 + j][r] + bv;
        if (OUT_F32) ((float*)Out)[(size_t)gm * EMB + gn] = v;
        else         ((_Float16*)Out)[(size_t)gm * EMB + gn] = (_Float16)v;
      }
    }
  }
}

// ---------------------------------------------------------------------------
// RoPE (concat-halves variant, matching the reference):
//   out[j]    = x[2j]*cos - x[2j+1]*sin     (j in [0,64))
//   out[j+64] = x[2j]*sin + x[2j+1]*cos
// One thread handles 16 pairs; out-of-place to avoid cross-thread RAW.
// outscale folds 1/sqrt(HDIM) into Q.
// ---------------------------------------------------------------------------
__global__ __launch_bounds__(256) void rope_f16(const _Float16* __restrict__ src,
                                                _Float16* __restrict__ dst,
                                                float outscale) {
  const int gid = blockIdx.x * blockDim.x + threadIdx.x;
  const int jb  = gid & 3;               // pair-block 0..3 (16 pairs each)
  const int h   = (gid >> 2) & (NHEAD - 1);
  const int s   = gid >> 6;
  const size_t base = (size_t)s * EMB + (size_t)h * HDIM;

  union { uint4 u[4]; _Float16 h[32]; } x;
  const uint4* sp = (const uint4*)(src + base + jb * 32);
  x.u[0] = sp[0]; x.u[1] = sp[1]; x.u[2] = sp[2]; x.u[3] = sp[3];

  union { uint4 u[2]; _Float16 h[16]; } o1, o2;
  const float lnth = 9.210340371976184f;  // ln(10000)
#pragma unroll
  for (int i = 0; i < 16; ++i) {
    const int j = jb * 16 + i;
    const float freq = __expf(-(2.0f * (float)j / (float)HDIM) * lnth);
    float sn, cs;
    __sincosf((float)s * freq, &sn, &cs);
    const float x1 = (float)x.h[2 * i];
    const float x2 = (float)x.h[2 * i + 1];
    o1.h[i] = (_Float16)((x1 * cs - x2 * sn) * outscale);
    o2.h[i] = (_Float16)((x1 * sn + x2 * cs) * outscale);
  }
  uint4* d1 = (uint4*)(dst + base + jb * 16);
  d1[0] = o1.u[0]; d1[1] = o1.u[1];
  uint4* d2 = (uint4*)(dst + base + 64 + jb * 16);
  d2[0] = o2.u[0]; d2[1] = o2.u[1];
}

// ---------------------------------------------------------------------------
// Attention with CAM mask, flash-style streaming softmax.
// Block = 128 threads (4 waves), one head per block, wave w owns queries
// [q0, q0+16). Key chunks of 32 over the allowed set only (45 chunks).
// K chunk: async global->LDS copy (ASYNCcnt), shared by all 4 waves.
// V chunk: cooperatively staged TRANSPOSED into LDS so PV B-fragments are
// contiguous ds_load_b128s. P (C layout) -> A layout via per-wave LDS tile
// (LDS ops are in-order within a wave).
// ---------------------------------------------------------------------------
__global__ __launch_bounds__(128) void attn_cam(const _Float16* __restrict__ Q,
                                                const _Float16* __restrict__ K,
                                                const _Float16* __restrict__ V,
                                                _Float16* __restrict__ O) {
  __shared__ __align__(16) _Float16 Vt[HDIM * LDSTRIDE];    // transposed V: [dim][key]
  __shared__ __align__(16) _Float16 Ks[32 * KSTRIDE];       // K chunk: [key][dim]
  __shared__ __align__(16) _Float16 Pt[4 * 16 * LDSTRIDE];  // per-wave P tiles [M][K]

  const int t    = threadIdx.x;
  const int wave = t >> 5;
  const int lane = t & 31;
  const int l16  = lane & 15;
  const int hi   = lane >> 4;
  const int head = blockIdx.y;
  const int q0   = blockIdx.x * 64 + wave * 16;
  const size_t hb = (size_t)head * HDIM;
  const int koffs = hi ? 8 : 0;

  // Resident Q fragments: 4 K-steps of 32 covering head dim 128.
  Frag16 qf[4];
  {
    const _Float16* qp = Q + (size_t)(q0 + l16) * EMB + hb + koffs;
#pragma unroll
    for (int kt = 0; kt < 4; ++kt) {
      qf[kt].u[0] = *(const uint4*)(qp + kt * 32);
      qf[kt].u[1] = *(const uint4*)(qp + kt * 32 + 16);
    }
  }

  v8f acc[8];
#pragma unroll
  for (int i = 0; i < 8; ++i) acc[i] = {};
  float mrow[8], lrow[8];
#pragma unroll
  for (int r = 0; r < 8; ++r) { mrow[r] = -1e30f; lrow[r] = 0.0f; }

  _Float16* pw = Pt + wave * 16 * LDSTRIDE;
  // Per-thread staging assignment: key row (0..31), 64-byte quarter (0..3).
  const int srow = t >> 2;
  const int squ  = t & 3;
  const unsigned ks_dst = (unsigned)(uintptr_t)(Ks + srow * KSTRIDE + squ * 32);

  for (int ci = 0; ci < NCHUNK; ++ci) {
    const int c0 = (ci < 13) ? ci * 32 : 3072 + (ci - 13) * 32;

    __syncthreads();   // everyone done reading Ks/Vt from the previous chunk

    // ---- async-stage K chunk (32 keys x 128 dims, row-major) into LDS ----
    async_copy64(ks_dst, (const void*)(K + (size_t)(c0 + srow) * EMB + hb + squ * 32));

    // ---- stage V chunk transposed: Vt[d][k] = V[c0+k][hb+d] ----
    {
      union { uint4 u[4]; _Float16 h[32]; } vx;
      const uint4* vp = (const uint4*)(V + (size_t)(c0 + srow) * EMB + hb + squ * 32);
      vx.u[0] = vp[0]; vx.u[1] = vp[1]; vx.u[2] = vp[2]; vx.u[3] = vp[3];
#pragma unroll
      for (int d = 0; d < 32; ++d)
        Vt[(squ * 32 + d) * LDSTRIDE + srow] = vx.h[d];
    }
    wait_async0();     // this wave's async K copies have landed in LDS
    __syncthreads();   // all waves' staging visible

    // S = Q @ K^T for 16 queries x 32 keys: two 16x16 C tiles, 4 K-steps.
    v8f s0 = {}, s1 = {};
#pragma unroll
    for (int kt = 0; kt < 4; ++kt) {
      Frag16 kf0, kf1;
      const _Float16* kp0 = Ks + l16 * KSTRIDE + kt * 32 + koffs;
      kf0.u[0] = *(const uint4*)kp0;
      kf0.u[1] = *(const uint4*)(kp0 + 16);
      const _Float16* kp1 = kp0 + 16 * KSTRIDE;
      kf1.u[0] = *(const uint4*)kp1;
      kf1.u[1] = *(const uint4*)(kp1 + 16);
      s0 = wmma_f16(qf[kt], kf0, s0);
      s1 = wmma_f16(qf[kt], kf1, s1);
    }

    // CAM mask (only the tail of chunk 12 is ever partially masked).
    const int key0 = c0 + l16;
    const int key1 = c0 + 16 + l16;
    if (key0 >= KEY_LO && key0 < KEY_HI) {
#pragma unroll
      for (int r = 0; r < 8; ++r) s0[r] = -1e30f;
    }
    if (key1 >= KEY_LO && key1 < KEY_HI) {
#pragma unroll
      for (int r = 0; r < 8; ++r) s1[r] = -1e30f;
    }

    // Online softmax. Row r of this wave's tile lives in VGPR slot r of the
    // lane half-group (lanes 0-15: rows 0-7; lanes 16-31: rows 8-15).
    float p0[8], p1[8];
#pragma unroll
    for (int r = 0; r < 8; ++r) {
      float mx = fmaxf(s0[r], s1[r]);
      for (int d = 1; d < 16; d <<= 1) mx = fmaxf(mx, __shfl_xor(mx, d, 16));
      const float mn = fmaxf(mrow[r], mx);
      const float sc = __expf(mrow[r] - mn);
      p0[r] = __expf(s0[r] - mn);
      p1[r] = __expf(s1[r] - mn);
      float rs = p0[r] + p1[r];
      for (int d = 1; d < 16; d <<= 1) rs += __shfl_xor(rs, d, 16);
      lrow[r] = lrow[r] * sc + rs;
      mrow[r] = mn;
#pragma unroll
      for (int dd = 0; dd < 8; ++dd) acc[dd][r] = acc[dd][r] * sc;
    }

    // P: C layout -> LDS [M][K] -> A-layout fragment (same-wave, in-order LDS).
#pragma unroll
    for (int r = 0; r < 8; ++r) {
      pw[(r + 8 * hi) * LDSTRIDE + l16]      = (_Float16)p0[r];
      pw[(r + 8 * hi) * LDSTRIDE + 16 + l16] = (_Float16)p1[r];
    }
    Frag16 pf;
    {
      const _Float16* pp = pw + l16 * LDSTRIDE + koffs;
      pf.u[0] = *(const uint4*)pp;
      pf.u[1] = *(const uint4*)(pp + 16);
    }

    // O += P @ V : 8 dim-subtiles, B fragments from transposed V in LDS.
#pragma unroll
    for (int ds = 0; ds < 8; ++ds) {
      Frag16 vf;
      const _Float16* vp = Vt + (ds * 16 + l16) * LDSTRIDE + koffs;
      vf.u[0] = *(const uint4*)vp;
      vf.u[1] = *(const uint4*)(vp + 16);
      acc[ds] = wmma_f16(pf, vf, acc[ds]);
    }
  }

  // Normalize and store f16 [s][e] for the output projection.
#pragma unroll
  for (int r = 0; r < 8; ++r) {
    const float inv = 1.0f / lrow[r];
    const int gm = q0 + r + 8 * hi;
#pragma unroll
    for (int ds = 0; ds < 8; ++ds)
      O[(size_t)gm * EMB + hb + ds * 16 + l16] = (_Float16)(acc[ds][r] * inv);
  }
}

// ---------------------------------------------------------------------------
extern "C" void kernel_launch(void* const* d_in, const int* in_sizes, int n_in,
                              void* d_out, int out_size, void* d_ws, size_t ws_size,
                              hipStream_t stream) {
  (void)in_sizes; (void)n_in; (void)out_size; (void)ws_size;

  const float* X  = (const float*)d_in[0];
  const float* Wq = (const float*)d_in[1];
  const float* bq = (const float*)d_in[2];
  const float* Wk = (const float*)d_in[3];
  const float* bk = (const float*)d_in[4];
  const float* Wv = (const float*)d_in[5];
  const float* bv = (const float*)d_in[6];
  const float* Wo = (const float*)d_in[7];
  const float* bo = (const float*)d_in[8];
  float* out = (float*)d_out;

  const size_t SZ = (size_t)SEQLEN * EMB;   // elements per f16 tensor
  _Float16* Qb = (_Float16*)d_ws;           // un-roped Q
  _Float16* Kb = Qb + SZ;                   // un-roped K
  _Float16* Vb = Kb + SZ;                   // V
  _Float16* Qr = Vb + SZ;                   // roped Q (scaled by 1/sqrt(128))
  _Float16* Kr = Qr + SZ;                   // roped K
  _Float16* AO = Qb;                        // attention out aliases Qb (dead after RoPE)

  const dim3 gg(EMB / 128, SEQLEN / 128);   // (16, 32)

  gemm_wmma<false, false><<<gg, 256, 0, stream>>>((const void*)X, Wq, bq, (void*)Qb);
  gemm_wmma<false, false><<<gg, 256, 0, stream>>>((const void*)X, Wk, bk, (void*)Kb);
  gemm_wmma<false, false><<<gg, 256, 0, stream>>>((const void*)X, Wv, bv, (void*)Vb);

  const dim3 rg((SEQLEN * NHEAD * 4) / 256);  // 1024 blocks
  const float qscale = 0.08838834764831845f;  // 1/sqrt(HDIM)
  rope_f16<<<rg, 256, 0, stream>>>(Qb, Qr, qscale);
  rope_f16<<<rg, 256, 0, stream>>>(Kb, Kr, 1.0f);

  attn_cam<<<dim3(SEQLEN / 64, NHEAD), 128, 0, stream>>>(Qr, Kr, Vb, AO);

  gemm_wmma<true, true><<<gg, 256, 0, stream>>>((const void*)AO, Wo, bo, (void*)out);
}
